// NeRFNetworkSecondDerivative_85100482003217
// MI455X (gfx1250) — matrix operand order repaired and checked by reading, not compiled, not run on hardware
//
#include <hip/hip_runtime.h>
#include <hip/hip_bf16.h>
#include <math.h>

typedef __attribute__((ext_vector_type(16))) _Float16 v16h;
typedef __attribute__((ext_vector_type(8)))  _Float16 v8h;
typedef __attribute__((ext_vector_type(8)))  float    v8f;

#define BNUM    1048576
#define TSZ     (1 << 19)
#define NTILES  (BNUM / 16)
#define WAVES   8
#define AST     72           // activation row stride (halves), 144B: 16B aligned + bank spread

// shared offsets in halves
#define OFF_SW0 0            // 64 x 40
#define OFF_SW1 2560         // 64 x 72
#define OFF_SW2 7168         // 16 x 72
#define OFF_CW0 8320         // 64 x 72
#define OFF_CW1 12928        // 64 x 72
#define OFF_CW2 17536        // 64 x 72
#define OFF_CW3 22144        // 16 x 72
#define OFF_ACT 23296        // 8 waves x 16 x 72
#define SMEM_HALVES 32512    // 65024 bytes

__constant__ float RES_TAB[16] = {
    16.f, 22.f, 30.f, 42.f, 58.f, 80.f, 111.f, 153.f,
    212.f, 294.f, 406.f, 561.f, 775.f, 1072.f, 1481.f, 2047.f};

// A-matrix 16x32 f16 fragment (ISA 7.12.2): lane m=lane&15, hi=lane>>4;
// V0..3 hold K=hi*8+0..7, V4..7 hold K=16+hi*8+0..7  -> two b128 LDS loads.
__device__ inline v16h load_afrag(const _Float16* act, int m, int hi, int kbase) {
    const _Float16* p = act + m * AST + kbase + hi * 8;
    v8h lo = *(const v8h*)p;
    v8h hh = *(const v8h*)(p + 16);
    v16h a;
#pragma unroll
    for (int i = 0; i < 8; ++i) { a[i] = lo[i]; a[8 + i] = hh[i]; }
    return a;
}

// B-matrix 32x16 f16 fragment: lane n=lane&15, kg=lane>>4;
// element i holds K=kbase+kg*8+i (i<8), K=kbase+16+kg*8+(i-8) (i>=8) of weight row (nbase+n).
__device__ inline v16h load_bfrag(const _Float16* w, int ws, int n, int kg,
                                  int nbase, int kbase) {
    const _Float16* p = w + (nbase + n) * ws + kbase + kg * 8;
    v8h lo = *(const v8h*)p;
    v8h hh = *(const v8h*)(p + 16);
    v16h b;
#pragma unroll
    for (int i = 0; i < 8; ++i) { b[i] = lo[i]; b[8 + i] = hh[i]; }
    return b;
}

__device__ inline v8f wmma16(v16h a, v16h b, v8f c) {
    return __builtin_amdgcn_wmma_f32_16x16x32_f16(false, a, false, b, (short)0, c,
                                                  false, false);
}

__device__ inline v8f vzero8() {
    v8f z;
#pragma unroll
    for (int i = 0; i < 8; ++i) z[i] = 0.f;
    return z;
}

// Dense layer: act(16xK) x W^T(KxN) + bias, optional relu, written back into act.
template <int KCH, int NT, bool RELU>
__device__ inline void layer(_Float16* act, const _Float16* w, int ws,
                             const float* bias, int lane) {
    const int m = lane & 15, hi = lane >> 4;
    v16h a[KCH];
#pragma unroll
    for (int c = 0; c < KCH; ++c) a[c] = load_afrag(act, m, hi, 32 * c);
    v8f acc[NT];
#pragma unroll
    for (int nt = 0; nt < NT; ++nt) {
        acc[nt] = vzero8();
#pragma unroll
        for (int c = 0; c < KCH; ++c)
            acc[nt] = wmma16(a[c], load_bfrag(w, ws, m, hi, nt * 16, 32 * c), acc[nt]);
    }
#pragma unroll
    for (int nt = 0; nt < NT; ++nt) {
        float bv = bias[nt * 16 + m];
#pragma unroll
        for (int v = 0; v < 8; ++v) {
            float val = acc[nt][v] + bv;
            if (RELU) val = fmaxf(val, 0.f);
            act[(v + 8 * hi) * AST + nt * 16 + m] = (_Float16)val;
        }
    }
}

__device__ inline void stage_w(_Float16* dst, const float* src, int srows,
                               int scols, int dstride, int drows, int tid) {
    const int total = drows * dstride;
    for (int i = tid; i < total; i += 256) {
        int r = i / dstride, c = i % dstride;
        float v = (r < srows && c < scols) ? src[r * scols + c] : 0.f;
        dst[i] = (_Float16)v;
    }
}

__global__ __launch_bounds__(256) void nerf_fused_kernel(
    const float* __restrict__ X, const float* __restrict__ D,
    const int* __restrict__ NN, const float2* __restrict__ GRID,
    const float* __restrict__ LAT,
    const float* __restrict__ sw0, const float* __restrict__ sb0,
    const float* __restrict__ sw1, const float* __restrict__ sb1,
    const float* __restrict__ sw2, const float* __restrict__ sb2,
    const float* __restrict__ cw0, const float* __restrict__ cb0,
    const float* __restrict__ cw1, const float* __restrict__ cb1,
    const float* __restrict__ cw2, const float* __restrict__ cb2,
    const float* __restrict__ cw3, const float* __restrict__ cb3,
    float* __restrict__ OUT) {
    __shared__ __align__(16) _Float16 sm[SMEM_HALVES];
    const int tid = threadIdx.x;
    const int lane = tid & 31, wave = tid >> 5;
    const int m = lane & 15, hi = lane >> 4;

    // One-time per-block f32 -> f16 weight staging with pad + bank-spread strides.
    stage_w(sm + OFF_SW0, sw0, 64, 32, 40, 64, tid);
    stage_w(sm + OFF_SW1, sw1, 64, 64, 72, 64, tid);
    stage_w(sm + OFF_SW2, sw2, 16, 64, 72, 16, tid);
    stage_w(sm + OFF_CW0, cw0, 64, 47, 72, 64, tid);   // pad 47 -> 64 inputs
    stage_w(sm + OFF_CW1, cw1, 64, 64, 72, 64, tid);
    stage_w(sm + OFF_CW2, cw2, 64, 64, 72, 64, tid);
    stage_w(sm + OFF_CW3, cw3, 3, 64, 72, 16, tid);    // pad 3 -> 16 outputs
    __syncthreads();

    _Float16* act = sm + OFF_ACT + wave * (16 * AST);

    for (int tile = blockIdx.x * WAVES + wave; tile < NTILES;
         tile += gridDim.x * WAVES) {
        const int pbase = tile * 16;
        const int p = pbase + m;  // each lane owns point m (both hi halves)

        // ---- hash-grid encode: lane handles 8 levels of its point ----
        const float x0 = X[3 * p], x1 = X[3 * p + 1], x2 = X[3 * p + 2];
        for (int j = 0; j < 8; ++j) {
            const int lvl = hi * 8 + j;
            const float res = RES_TAB[lvl];
            float px = x0 * res, py = x1 * res, pz = x2 * res;
            float fx = floorf(px), fy = floorf(py), fz = floorf(pz);
            float wx = px - fx, wy = py - fy, wz = pz - fz;
            unsigned ix = (unsigned)(int)fx, iy = (unsigned)(int)fy,
                     iz = (unsigned)(int)fz;
            const float2* g = GRID + (size_t)lvl * TSZ;
            float f0 = 0.f, f1 = 0.f;
#pragma unroll
            for (int c = 0; c < 8; ++c) {
                unsigned cx = c & 1, cy = (c >> 1) & 1, cz = (c >> 2) & 1;
                unsigned h = (ix + cx) ^ ((iy + cy) * 2654435761u) ^
                             ((iz + cz) * 805459861u);
                h &= (TSZ - 1);
                float2 gv = g[h];
                float wc = (cx ? wx : 1.f - wx) * (cy ? wy : 1.f - wy) *
                           (cz ? wz : 1.f - wz);
                f0 += gv.x * wc;
                f1 += gv.y * wc;
            }
            act[m * AST + 2 * lvl] = (_Float16)f0;
            act[m * AST + 2 * lvl + 1] = (_Float16)f1;
        }

        // ---- sigma MLP: 32 -> 64 -> 64 -> 16 ----
        layer<1, 4, true>(act, sm + OFF_SW0, 40, sb0, lane);
        layer<2, 4, true>(act, sm + OFF_SW1, 72, sb1, lane);

        v8f s;
        {
            v16h a0 = load_afrag(act, m, hi, 0);
            v16h a1 = load_afrag(act, m, hi, 32);
            s = vzero8();
            s = wmma16(a0, load_bfrag(sm + OFF_SW2, 72, m, hi, 0, 0), s);
            s = wmma16(a1, load_bfrag(sm + OFF_SW2, 72, m, hi, 0, 32), s);
        }
        // sigma out (col 0) + geo (cols 1..15 -> cin cols 0..14)
        {
            float bv = sb2[m];
#pragma unroll
            for (int v = 0; v < 8; ++v) {
                float val = s[v] + bv;
                int row = v + 8 * hi;
                if (m == 0)
                    OUT[pbase + row] = val;
                else
                    act[row * AST + (m - 1)] = (_Float16)val;
            }
        }

        // ---- build color input: [geo(15) | sh16(16) | latent(16) | 0-pad] ----
        if (hi == 0) {  // SH basis for point m  (dd == d exactly)
            float x = D[3 * p], y = D[3 * p + 1], z = D[3 * p + 2];
            float xx = x * x, yy = y * y, zz = z * z;
            float xy = x * y, yz = y * z, xz = x * z;
            float sh[16];
            sh[0] = 0.28209479177387814f;
            sh[1] = -0.48860251190291987f * y;
            sh[2] = 0.48860251190291987f * z;
            sh[3] = -0.48860251190291987f * x;
            sh[4] = 1.0925484305920792f * xy;
            sh[5] = -1.0925484305920792f * yz;
            sh[6] = 0.94617469575756f * zz - 0.31539156525252f;
            sh[7] = -1.0925484305920792f * xz;
            sh[8] = 0.5462742152960396f * (xx - yy);
            sh[9] = 0.5900435899266435f * y * (-3.f * xx + yy);
            sh[10] = 2.890611442640554f * xy * z;
            sh[11] = 0.4570457994644657f * y * (1.f - 5.f * zz);
            sh[12] = 0.3731763325901154f * z * (5.f * zz - 3.f);
            sh[13] = 0.4570457994644657f * x * (1.f - 5.f * zz);
            sh[14] = 1.445305721320277f * z * (xx - yy);
            sh[15] = 0.5900435899266435f * x * (-xx + 3.f * yy);
#pragma unroll
            for (int j = 0; j < 16; ++j)
                act[m * AST + 15 + j] = (_Float16)sh[j];
        } else {  // latent gather for point m
            const float* le = LAT + (size_t)NN[p] * 16;
#pragma unroll
            for (int j = 0; j < 16; ++j)
                act[m * AST + 31 + j] = (_Float16)le[j];
        }
        {  // zero pad cols 47..63 (split across lane halves)
            int c0 = hi ? 56 : 47, c1 = hi ? 64 : 56;
            for (int c = c0; c < c1; ++c) act[m * AST + c] = (_Float16)0.f;
        }

        // ---- color MLP: 64 -> 64 -> 64 -> 64 -> 3 ----
        layer<2, 4, true>(act, sm + OFF_CW0, 72, cb0, lane);
        layer<2, 4, true>(act, sm + OFF_CW1, 72, cb1, lane);
        layer<2, 4, true>(act, sm + OFF_CW2, 72, cb2, lane);
        {
            v16h a0 = load_afrag(act, m, hi, 0);
            v16h a1 = load_afrag(act, m, hi, 32);
            v8f cc = vzero8();
            cc = wmma16(a0, load_bfrag(sm + OFF_CW3, 72, m, hi, 0, 0), cc);
            cc = wmma16(a1, load_bfrag(sm + OFF_CW3, 72, m, hi, 0, 32), cc);
            if (m < 3) {
                float bv = cb3[m];
#pragma unroll
                for (int v = 0; v < 8; ++v) {
                    float val = cc[v] + bv;
                    val = 1.f / (1.f + expf(-val));
                    OUT[BNUM + (size_t)(pbase + v + 8 * hi) * 3 + m] = val;
                }
            }
        }
    }
}

extern "C" void kernel_launch(void* const* d_in, const int* in_sizes, int n_in,
                              void* d_out, int out_size, void* d_ws, size_t ws_size,
                              hipStream_t stream) {
    (void)in_sizes; (void)n_in; (void)out_size; (void)d_ws; (void)ws_size;
    nerf_fused_kernel<<<2048, 256, 0, stream>>>(
        (const float*)d_in[0],   // x
        (const float*)d_in[1],   // d
        (const int*)d_in[2],     // n
        (const float2*)d_in[3],  // grid
        (const float*)d_in[4],   // latent_emb
        (const float*)d_in[5], (const float*)d_in[6],    // sw0, sb0
        (const float*)d_in[7], (const float*)d_in[8],    // sw1, sb1
        (const float*)d_in[9], (const float*)d_in[10],   // sw2, sb2
        (const float*)d_in[11], (const float*)d_in[12],  // cw0, cb0
        (const float*)d_in[13], (const float*)d_in[14],  // cw1, cb1
        (const float*)d_in[15], (const float*)d_in[16],  // cw2, cb2
        (const float*)d_in[17], (const float*)d_in[18],  // cw3, cb3
        (float*)d_out);
}
